// NeuralECMModel_60705067762111
// MI455X (gfx1250) — matrix operand order repaired and checked by reading, not compiled, not run on hardware
//
#include <hip/hip_runtime.h>
#include <math.h>

typedef __attribute__((ext_vector_type(16))) _Float16 v16h;
typedef __attribute__((ext_vector_type(8)))  float    v8f;

#define N_NODES 20000
#define KNBR    63
#define KP1     64
#define DDIM    50
#define EDIM    300

#define TILE_NODES 64      // nodes per block
#define BLOCK_T    128     // 4 waves * 32 lanes; each wave owns a 16-node WMMA tile

#define KC_ENT  10         // ceil(300/32) k-chunks for ent projection
#define KBIL    2500       // 50*50 flattened bilinear contraction dim
#define KC_BIL  79         // ceil(2500/32)
#define KC_AGG  2          // ceil(50/32)
#define OT      4          // ceil(50/16) output column tiles (padded to 64)

// d_ws layout (halfs): packed B-fragment copies of the three weight matrices
#define WPH_OFF   0
#define WPH_CNT   (KC_ENT * OT * 32 * 16)    // 20480 halfs
#define WBILH_OFF (WPH_OFF + WPH_CNT)
#define WBILH_CNT (KC_BIL * OT * 32 * 16)    // 161792 halfs
#define WGH_OFF   (WBILH_OFF + WBILH_CNT)
#define WGH_CNT   (KC_AGG * OT * 32 * 16)    // 4096 halfs
#define WS_HALFS  (WGH_OFF + WGH_CNT)        // 186368 halfs = 364 KB

__device__ __forceinline__ v8f wmma16x16x32f16(v16h a, v16h b, v8f c) {
    // D = A(16x32 f16) * B(32x16 f16) + C(16x16 f32)
    return __builtin_amdgcn_wmma_f32_16x16x32_f16(
        /*neg_a=*/false, a, /*neg_b=*/false, b,
        /*c_mod=*/(short)0, c, /*reuse_a=*/false, /*reuse_b=*/false);
}

// ---------------------------------------------------------------------------
// Prep: pack W[o*ld + k] (f32, o<ODIM, k<KDIM) into f16 B-fragments laid out as
//   frag(kc, ot, lane)[e] = W[(ot*16 + lane%16) * ld + kc*32 + (lane<16?0:8) + e + (e<8?0:8)]
// so the main loop loads one aligned v16h (32 B) per lane per WMMA. Out-of-range
// (k,o) are zero-filled, implementing the K/N padding for free.
// ---------------------------------------------------------------------------
__global__ void pack_frag_kernel(const float* __restrict__ W, _Float16* __restrict__ outp,
                                 int KC, int KDIM, int ODIM, int ld)
{
    int gid = blockIdx.x * blockDim.x + threadIdx.x;   // one fragment (16 halfs) per thread
    int total = KC * OT * 32;
    if (gid >= total) return;
    int lane = gid & 31;
    int ot   = (gid >> 5) & (OT - 1);
    int kc   = gid >> 7;
    int o     = ot * 16 + (lane & 15);
    int kbase = kc * 32 + ((lane < 16) ? 0 : 8);
    _Float16* dst = outp + (size_t)gid * 16;
    #pragma unroll
    for (int e = 0; e < 16; ++e) {
        int k = kbase + e + ((e < 8) ? 0 : 8);
        float v = (k < KDIM && o < ODIM) ? W[(size_t)o * ld + k] : 0.0f;
        dst[e] = (_Float16)v;
    }
}

__device__ __forceinline__ v16h load_bfrag(const _Float16* __restrict__ WH,
                                           int kc, int ot, int lane) {
    return *(const v16h*)(WH + ((size_t)(kc * OT + ot) * 32 + lane) * 16);
}

// ---------------------------------------------------------------------------
// Fused ECM kernel: 4 waves/block, 16 nodes per wave, 64 nodes per block.
// ---------------------------------------------------------------------------
__global__ __launch_bounds__(BLOCK_T)
void ecm_fused_kernel(const float* __restrict__ q,        // [N,50]
                      const float* __restrict__ ent_in,   // [N,300]
                      const float* __restrict__ nb,       // [N,63,50]
                      const float* __restrict__ sc,       // [N,64]
                      const _Float16* __restrict__ WpH,   // packed fragments
                      const float* __restrict__ bp,       // [50]
                      const _Float16* __restrict__ WbilH, // packed fragments
                      const float* __restrict__ bbil,     // [50]
                      const _Float16* __restrict__ WgH,   // packed fragments
                      const float* __restrict__ gb,       // [50]
                      const float* __restrict__ Wr,       // [1,50]
                      const float* __restrict__ br,       // [1]
                      float* __restrict__ out)            // [N]
{
    __shared__ float q_lds[TILE_NODES][DDIM];     // 12.8 KB
    __shared__ float ent_lds[TILE_NODES][64];     // 16.0 KB (cols 50..63 zeroed)
    __shared__ float node_lds[TILE_NODES][DDIM];  // 12.8 KB
    __shared__ float wsum_lds[TILE_NODES][64];    // 16.0 KB (cols 50..63 zeroed)
    __shared__ float out_acc[TILE_NODES];         // 0.25 KB

    const int tid  = threadIdx.x;
    const int lane = tid & 31;
    const int wv   = tid >> 5;
    const int n0   = blockIdx.x * TILE_NODES;

    if (tid < TILE_NODES) out_acc[tid] = 0.0f;

    // cooperative coalesced load of the query tile (node index clamped; store guarded)
    for (int idx = tid; idx < TILE_NODES * DDIM; idx += BLOCK_T) {
        int ml = idx / DDIM, d = idx - ml * DDIM;
        int n = min(n0 + ml, N_NODES - 1);
        q_lds[ml][d] = q[(size_t)n * DDIM + d];
    }
    __syncthreads();

    // WMMA per-lane fragment geometry (16-bit A 16x32, B 32x16, f32 C/D; wave32)
    const int mloc   = wv * 16 + (lane & 15);            // A-row this lane feeds (local node)
    const int nglob  = min(n0 + mloc, N_NODES - 1);
    const int kbaseA = (lane < 16) ? 0 : 8;
    const int colB   = lane & 15;

    // ---------------- Phase A: ent = entity_emb @ Wp^T + bp  (K=300, WMMA) -----------
    v8f accA[OT];
    #pragma unroll
    for (int t = 0; t < OT; ++t)
        #pragma unroll
        for (int r = 0; r < 8; ++r) accA[t][r] = 0.0f;

    for (int kc = 0; kc < KC_ENT; ++kc) {
        v16h a;
        #pragma unroll
        for (int e = 0; e < 16; ++e) {
            int kl = (e < 8) ? (kbaseA + e) : (16 + kbaseA + (e - 8));
            int k = kc * 32 + kl;
            float v = (k < EDIM) ? ent_in[(size_t)nglob * EDIM + k] : 0.0f;
            a[e] = (_Float16)v;
        }
        #pragma unroll
        for (int t = 0; t < OT; ++t) {
            v16h b = load_bfrag(WpH, kc, t, lane);
            accA[t] = wmma16x16x32f16(a, b, accA[t]);
        }
    }
    #pragma unroll
    for (int t = 0; t < OT; ++t) {
        int o = t * 16 + colB;
        float bias = (o < DDIM) ? bp[o] : 0.0f;
        #pragma unroll
        for (int r = 0; r < 8; ++r) {
            int m = wv * 16 + ((lane < 16) ? r : r + 8);
            ent_lds[m][o] = accA[t][r] + bias;   // o>=50 stays exactly 0 (B was zero-padded)
        }
    }
    __syncthreads();

    // ---------------- Phase B: node = (q (x) ent) @ WbilR^T + bbil  (K=2500, WMMA) ---
    // A row built on the fly: t[m, i*50+j] = q[m,i]*ent[m,j]; B fragments pre-packed f16
    v8f accB[OT];
    #pragma unroll
    for (int t = 0; t < OT; ++t)
        #pragma unroll
        for (int r = 0; r < 8; ++r) accB[t][r] = 0.0f;

    for (int kc = 0; kc < KC_BIL; ++kc) {
        v16h a;
        #pragma unroll
        for (int e = 0; e < 16; ++e) {
            int kl = (e < 8) ? (kbaseA + e) : (16 + kbaseA + (e - 8));
            int k = kc * 32 + kl;
            float v = 0.0f;
            if (k < KBIL) {
                int i = k / DDIM;
                int j = k - i * DDIM;
                v = q_lds[mloc][i] * ent_lds[mloc][j];
            }
            a[e] = (_Float16)v;
        }
        #pragma unroll
        for (int t = 0; t < OT; ++t) {
            v16h b = load_bfrag(WbilH, kc, t, lane);
            accB[t] = wmma16x16x32f16(a, b, accB[t]);
        }
    }
    #pragma unroll
    for (int t = 0; t < OT; ++t) {
        int o = t * 16 + colB;
        if (o < DDIM) {
            float bias = bbil[o];
            #pragma unroll
            for (int r = 0; r < 8; ++r) {
                int m = wv * 16 + ((lane < 16) ? r : r + 8);
                node_lds[m][o] = accB[t][r] + bias;
            }
        }
    }
    __syncthreads();

    // ---------------- Phase C: bandwidth phase — score-weighted neighbor sum ---------
    // lane owns dims d0=lane (0..31) and d1=lane+32 (32..63); contiguous segments per row
    for (int ml = 0; ml < 16; ++ml) {
        int m = wv * 16 + ml;
        int n = min(n0 + m, N_NODES - 1);
        const float* srow = sc + (size_t)n * KP1;
        const float* nrow = nb + (size_t)n * KNBR * DDIM;
        int d1 = lane + 32;
        float a0 = 0.0f, a1 = 0.0f;
        #pragma unroll 7
        for (int k = 0; k < KNBR; ++k) {
            float s = srow[k];
            a0 = fmaf(s, nrow[k * DDIM + lane], a0);
            if (d1 < DDIM) a1 = fmaf(s, nrow[k * DDIM + d1], a1);
        }
        float s63 = srow[KNBR];
        a0 = fmaf(s63, node_lds[m][lane], a0);
        if (d1 < DDIM) a1 = fmaf(s63, node_lds[m][d1], a1);
        wsum_lds[m][lane] = a0;
        wsum_lds[m][d1]   = (d1 < DDIM) ? a1 : 0.0f;   // zero-pads cols 50..63
    }
    __syncthreads();

    // ---------------- Phase D: agg = wsum @ Wg^T ; ELU ; rank = feats . Wr + br ------
    v8f accD[OT];
    #pragma unroll
    for (int t = 0; t < OT; ++t)
        #pragma unroll
        for (int r = 0; r < 8; ++r) accD[t][r] = 0.0f;

    #pragma unroll
    for (int kc = 0; kc < KC_AGG; ++kc) {
        v16h a;
        #pragma unroll
        for (int e = 0; e < 16; ++e) {
            int kl = (e < 8) ? (kbaseA + e) : (16 + kbaseA + (e - 8));
            int k = kc * 32 + kl;                 // k < 64 always; cols>=50 already zero
            a[e] = (_Float16)wsum_lds[mloc][k];
        }
        #pragma unroll
        for (int t = 0; t < OT; ++t) {
            v16h b = load_bfrag(WgH, kc, t, lane);
            accD[t] = wmma16x16x32f16(a, b, accD[t]);
        }
    }
    #pragma unroll
    for (int t = 0; t < OT; ++t) {
        int o = t * 16 + colB;
        float wr    = (o < DDIM) ? Wr[o] : 0.0f;
        float gbias = (o < DDIM) ? gb[o] : 0.0f;
        #pragma unroll
        for (int r = 0; r < 8; ++r) {
            int m = wv * 16 + ((lane < 16) ? r : r + 8);
            float x = accD[t][r] + gbias;
            float f = (x > 0.0f) ? x : (__expf(x) - 1.0f);   // ELU
            atomicAdd(&out_acc[m], f * wr);                   // ds_add_f32
        }
    }
    __syncthreads();

    if (tid < TILE_NODES) {
        int n = n0 + tid;
        if (n < N_NODES) out[n] = out_acc[tid] + br[0];
    }
}

extern "C" void kernel_launch(void* const* d_in, const int* in_sizes, int n_in,
                              void* d_out, int out_size, void* d_ws, size_t ws_size,
                              hipStream_t stream) {
    (void)in_sizes; (void)n_in; (void)out_size; (void)ws_size;
    const float* q      = (const float*)d_in[0];   // query_emb      [N,50]
    const float* ent_in = (const float*)d_in[1];   // entity_emb     [N,300]
    const float* nb     = (const float*)d_in[2];   // neighbor_embs  [N,63,50]
    const float* sc     = (const float*)d_in[3];   // neighbor_scores[N,64]
    const float* Wp     = (const float*)d_in[4];   // [50,300]
    const float* bp     = (const float*)d_in[5];   // [50]
    const float* Wbil   = (const float*)d_in[6];   // [50,50,50]
    const float* bbil   = (const float*)d_in[7];   // [50]
    const float* Wg     = (const float*)d_in[8];   // [50,50]
    const float* gb     = (const float*)d_in[9];   // [50]
    const float* Wr     = (const float*)d_in[10];  // [1,50]
    const float* br     = (const float*)d_in[11];  // [1]
    float* out = (float*)d_out;

    _Float16* wsH   = (_Float16*)d_ws;             // needs WS_HALFS*2 = 364 KB of scratch
    _Float16* WpH   = wsH + WPH_OFF;
    _Float16* WbilH = wsH + WBILH_OFF;
    _Float16* WgH   = wsH + WGH_OFF;

    // Pack weights into f16 WMMA B-fragment layout (deterministic; runs every call).
    {
        int tWp   = KC_ENT * OT * 32;   // 1280 fragments
        int tWbil = KC_BIL * OT * 32;   // 10112 fragments
        int tWg   = KC_AGG * OT * 32;   // 256 fragments
        pack_frag_kernel<<<(tWp   + 255) / 256, 256, 0, stream>>>(Wp,   WpH,   KC_ENT, EDIM, DDIM, EDIM);
        pack_frag_kernel<<<(tWbil + 255) / 256, 256, 0, stream>>>(Wbil, WbilH, KC_BIL, KBIL, DDIM, KBIL);
        pack_frag_kernel<<<(tWg   + 255) / 256, 256, 0, stream>>>(Wg,   WgH,   KC_AGG, DDIM, DDIM, DDIM);
    }

    int blocks = (N_NODES + TILE_NODES - 1) / TILE_NODES;  // 313
    ecm_fused_kernel<<<blocks, BLOCK_T, 0, stream>>>(
        q, ent_in, nb, sc, WpH, bp, WbilH, bbil, WgH, gb, Wr, br, out);
}
